// BasePkaGNN_88914412961908
// MI455X (gfx1250) — matrix operand order repaired and checked by reading, not compiled, not run on hardware
//
#include <hip/hip_runtime.h>

typedef __attribute__((ext_vector_type(16))) __bf16 v16bf;
typedef __attribute__((ext_vector_type(8)))  float  v8f;

#define HDIM  300
#define BONDD 11
#define KPOS  8
#define KP_H  320   // padded K for 300/311-wide inputs (mult of 32)
#define KP_CAT 608  // padded K for 600-wide concat (19*32)
#define NP_H  320   // padded cols for 300-col weights (mult of 64)
#define NP_C  128   // padded cols for 128-col weights

static __device__ __forceinline__ __bf16 f2b(float f) { return (__bf16)f; }

// ---------------------------------------------------------------- utility
__global__ void k_zero_f(float* p, long n) {
  long i = (long)blockIdx.x * blockDim.x + threadIdx.x;
  if (i < n) p[i] = 0.f;
}
__global__ void k_copy_f(float* d, const float* s, long n) {
  long i = (long)blockIdx.x * blockDim.x + threadIdx.x;
  if (i < n) d[i] = s[i];
}

// ------------------------------------------------------------- bf16 packs
// W[K,Nw] (row-major f32) -> Wp[Kp,Np] bf16, zero padded
__global__ void k_pack_w(__bf16* Wp, const float* W, int K, int Nw, int Kp, int Np) {
  long i = (long)blockIdx.x * blockDim.x + threadIdx.x;
  long tot = (long)Kp * Np;
  if (i >= tot) return;
  int k = (int)(i / Np), n = (int)(i % Np);
  float v = (k < K && n < Nw) ? W[(long)k * Nw + n] : 0.f;
  Wp[i] = f2b(v);
}
// A row e = [x[src[e]] (300) | ea[e] (11) | 0-pad] -> [E, 320] bf16
__global__ void k_pack_edge(__bf16* Ap, const float* x, const float* ea,
                            const int* src, long E) {
  long i = (long)blockIdx.x * blockDim.x + threadIdx.x;
  long tot = E * KP_H;
  if (i >= tot) return;
  long e = i / KP_H; int c = (int)(i % KP_H);
  float v = 0.f;
  if (c < HDIM)            v = x[(long)src[e] * HDIM + c];
  else if (c < HDIM+BONDD) v = ea[e * BONDD + (c - HDIM)];
  Ap[i] = f2b(v);
}
// A row e = a[src[e]] - h[rev[e]] -> [E, 320] bf16
__global__ void k_pack_msg(__bf16* Ap, const float* agg, const float* h,
                           const int* src, const int* rev, long E) {
  long i = (long)blockIdx.x * blockDim.x + threadIdx.x;
  long tot = E * KP_H;
  if (i >= tot) return;
  long e = i / KP_H; int c = (int)(i % KP_H);
  float v = 0.f;
  if (c < HDIM) v = agg[(long)src[e] * HDIM + c] - h[(long)rev[e] * HDIM + c];
  Ap[i] = f2b(v);
}
// A row n = [x[n] (300) | agg[n] (300) | pad] -> [M, 608] bf16
__global__ void k_pack_cat2(__bf16* Ap, const float* x, const float* agg, long M) {
  long i = (long)blockIdx.x * blockDim.x + threadIdx.x;
  long tot = M * KP_CAT;
  if (i >= tot) return;
  long m = i / KP_CAT; int c = (int)(i % KP_CAT);
  float v = 0.f;
  if (c < HDIM)        v = x[m * HDIM + c];
  else if (c < 2*HDIM) v = agg[m * HDIM + (c - HDIM)];
  Ap[i] = f2b(v);
}
// A row m = [X[m] (300) | pad] -> [M, 320] bf16
__global__ void k_pack_plain(__bf16* Ap, const float* X, long M) {
  long i = (long)blockIdx.x * blockDim.x + threadIdx.x;
  long tot = M * KP_H;
  if (i >= tot) return;
  long m = i / KP_H; int c = (int)(i % KP_H);
  Ap[i] = f2b(c < HDIM ? X[m * HDIM + c] : 0.f);
}

// ---------------------------------------------- segment_sum via f32 atomics
__global__ void k_scatter_add(float* agg, const float* h, const int* dst, long E) {
  long i = (long)blockIdx.x * blockDim.x + threadIdx.x;
  long tot = E * HDIM;
  if (i >= tot) return;
  long e = i / HDIM; int c = (int)(i % HDIM);
  atomicAdd(&agg[(long)dst[e] * HDIM + c], h[i]);
}

// ---------------------------------------------------------- WMMA bf16 GEMM
// C[M,Nout] = act(A[M,lda]*B[lda? no: Kp x ldb] + bias [+ addend]),
// 128 threads = 4 waves, 64x64 C tile, K-step 32, LDS staged.
__launch_bounds__(128)
__global__ void k_gemm(const __bf16* __restrict__ A, int lda,
                       const __bf16* __restrict__ B, int ldb,
                       const float* __restrict__ bias,
                       const float* __restrict__ addend,
                       float* __restrict__ C,
                       int M, int Kp, int Nout, int relu) {
  __shared__ __bf16 As[64][32];
  __shared__ __bf16 Bs[4][32][16];  // [n-subtile][lane][frag element] pre-swizzled
  const int tid  = threadIdx.x;
  const int wave = tid >> 5;
  const int lane = tid & 31;
  const int m0   = blockIdx.x * 64;
  const int n0   = blockIdx.y * 64;
  const int r    = lane & 15;
  const int hi   = lane >> 4;

  v8f acc0 = {}, acc1 = {}, acc2 = {}, acc3 = {};

  for (int k0 = 0; k0 < Kp; k0 += 32) {
    // stage A tile 64x32: each thread copies 16 contiguous bf16
    {
      int ar = tid >> 1, ac = (tid & 1) * 16;
      long gr = (long)m0 + ar;
      if (gr < M) {
        const __bf16* ap = A + gr * (long)lda + k0 + ac;
#pragma unroll
        for (int j = 0; j < 16; ++j) As[ar][ac + j] = ap[j];
      } else {
#pragma unroll
        for (int j = 0; j < 16; ++j) As[ar][ac + j] = f2b(0.f);
      }
    }
    // stage B tile 32x64 pre-swizzled into per-lane fragment order
    {
#pragma unroll
      for (int q = 0; q < 16; ++q) {
        int idx = tid * 16 + q;          // 0..2047
        int kk = idx >> 6, n = idx & 63; // K row, N col within tile
        Bs[n >> 4][(n & 15) | ((kk & 8) << 1)][(kk & 7) | ((kk & 16) >> 1)] =
            B[(long)(k0 + kk) * ldb + n0 + n];
      }
    }
    __syncthreads();

    // build A fragment (16x32) for this wave's 16 rows
    v16bf af;
    const __bf16* ap = &As[wave * 16 + r][hi * 8];
#pragma unroll
    for (int j = 0; j < 8; ++j) { af[j] = ap[j]; af[j + 8] = ap[16 + j]; }

    // 4 B fragments (32x16 each), contiguous per lane in LDS
    v16bf b0, b1, b2, b3;
#pragma unroll
    for (int j = 0; j < 16; ++j) {
      b0[j] = Bs[0][lane][j]; b1[j] = Bs[1][lane][j];
      b2[j] = Bs[2][lane][j]; b3[j] = Bs[3][lane][j];
    }
    acc0 = __builtin_amdgcn_wmma_f32_16x16x32_bf16(false, af, false, b0, (short)0, acc0, false, false);
    acc1 = __builtin_amdgcn_wmma_f32_16x16x32_bf16(false, af, false, b1, (short)0, acc1, false, false);
    acc2 = __builtin_amdgcn_wmma_f32_16x16x32_bf16(false, af, false, b2, (short)0, acc2, false, false);
    acc3 = __builtin_amdgcn_wmma_f32_16x16x32_bf16(false, af, false, b3, (short)0, acc3, false, false);
    __syncthreads();
  }

  // epilogue: lane col = r within subtile, element i -> row i + 8*hi
#pragma unroll
  for (int t = 0; t < 4; ++t) {
    v8f a = (t == 0) ? acc0 : (t == 1) ? acc1 : (t == 2) ? acc2 : acc3;
    int gn = n0 + t * 16 + r;
    if (gn >= Nout) continue;
    float bs = bias ? bias[gn] : 0.f;
#pragma unroll
    for (int i = 0; i < 8; ++i) {
      int gm = m0 + wave * 16 + i + 8 * hi;
      if (gm < M) {
        float v = a[i] + bs;
        if (addend) v += addend[(long)gm * Nout + gn];
        if (relu) v = v > 0.f ? v : 0.f;
        C[(long)gm * Nout + gn] = v;
      }
    }
  }
}

// ------------------------------------------------------------- scan pieces
// Classifier stage 2 (128->2), log_softmax, weighted CE, block-reduced sum.
__global__ void k_ce(const float* __restrict__ f1, const float* __restrict__ Wc2,
                     const float* __restrict__ bc2, const int* idxs, int step,
                     float ratio, long N, float* lc_accum, float* logits_out) {
  __shared__ float wc2s[256];
  __shared__ float red[256];
  int tid = threadIdx.x;
  wc2s[tid] = Wc2[tid];
  __syncthreads();
  long n = (long)blockIdx.x * blockDim.x + tid;
  float nll = 0.f;
  if (n < N) {
    const float* f = f1 + n * 128;
    float l0 = bc2[0], l1 = bc2[1];
#pragma unroll 4
    for (int j = 0; j < 128; ++j) {
      float fj = f[j];
      l0 += fj * wc2s[j * 2];
      l1 += fj * wc2s[j * 2 + 1];
    }
    float m = fmaxf(l0, l1);
    float lse = m + logf(expf(l0 - m) + expf(l1 - m));
    int idx = idxs[step];
    nll = (n == (long)idx) ? (-ratio * (l1 - lse)) : (-(l0 - lse));
    if (logits_out) { logits_out[n * 2] = l0; logits_out[n * 2 + 1] = l1; }
  }
  red[tid] = nll;
  __syncthreads();
  for (int s = 128; s > 0; s >>= 1) {
    if (tid < s) red[tid] += red[tid + s];
    __syncthreads();
  }
  if (tid == 0) atomicAdd(lc_accum, red[0]);
}

// Regressor for one row (300->128 relu ->1), normalized MSE vs label.
__global__ void k_reg_row(const float* h_cur, const int* idxs, int step,
                          const float* Wr1, const float* br1,
                          const float* Wr2, const float* br2,
                          const float* labels, const float* pka_std, float* lr) {
  __shared__ float hi[HDIM];
  __shared__ float f[128];
  int tid = threadIdx.x;
  int idx = idxs[step];
  for (int j = tid; j < HDIM; j += 128) hi[j] = h_cur[(long)idx * HDIM + j];
  __syncthreads();
  float s = br1[tid];
  for (int j = 0; j < HDIM; ++j) s += hi[j] * Wr1[(long)j * 128 + tid];
  f[tid] = s > 0.f ? s : 0.f;
  __syncthreads();
  if (tid == 0) {
    float pred = br2[0];
    for (int t = 0; t < 128; ++t) pred += f[t] * Wr2[t];
    float d = (pred - labels[idx]) / pka_std[0];
    lr[step] = d * d;
  }
}

// h_in[idx] = hi * tanh(hi@Wg + bg) + hi  (h_in already holds h_static copy)
__global__ void k_gate_row(float* h_in, const float* h_cur, const int* idxs,
                           int step, const float* Wg, const float* bg) {
  __shared__ float hi[HDIM];
  int tid = threadIdx.x;
  int idx = idxs[step];
  if (tid < HDIM) hi[tid] = h_cur[(long)idx * HDIM + tid];
  __syncthreads();
  if (tid < HDIM) {
    float g = bg[tid];
    for (int j = 0; j < HDIM; ++j) g += hi[j] * Wg[(long)j * HDIM + tid];
    h_in[(long)idx * HDIM + tid] = hi[tid] * tanhf(g) + hi[tid];
  }
}

// Final regressor stage 2 over all nodes.
__global__ void k_pka(const float* __restrict__ f1, const float* __restrict__ Wr2,
                      const float* __restrict__ br2, float* out, long N) {
  __shared__ float w[128];
  if (threadIdx.x < 128) w[threadIdx.x] = Wr2[threadIdx.x];
  __syncthreads();
  long n = (long)blockIdx.x * blockDim.x + threadIdx.x;
  if (n < N) {
    const float* f = f1 + n * 128;
    float s = br2[0];
#pragma unroll 4
    for (int j = 0; j < 128; ++j) s += f[j] * w[j];
    out[n] = s;
  }
}

// Find the 8 positive-labelled atoms, sorted ascending by label value.
__global__ void k_topk(const float* labels, long N, int* idx_out) {
  __shared__ int cnt;
  __shared__ int si[32];
  __shared__ float sv[32];
  if (threadIdx.x == 0) cnt = 0;
  __syncthreads();
  for (long n = threadIdx.x; n < N; n += blockDim.x) {
    float v = labels[n];
    if (v > 0.f) {
      int p = atomicAdd(&cnt, 1);
      if (p < 32) { si[p] = (int)n; sv[p] = v; }
    }
  }
  __syncthreads();
  if (threadIdx.x == 0) {
    int m = cnt < 32 ? cnt : 32;
    for (int i = 1; i < m; ++i) {       // insertion sort ascending
      float v = sv[i]; int ix = si[i]; int j = i - 1;
      while (j >= 0 && sv[j] > v) { sv[j + 1] = sv[j]; si[j + 1] = si[j]; --j; }
      sv[j + 1] = v; si[j + 1] = ix;
    }
    int base = m >= KPOS ? m - KPOS : 0; // keep 8 largest, ascending
    for (int i = 0; i < KPOS; ++i) {
      int p = base + i;
      idx_out[i] = (p < m) ? si[p] : 0;
    }
  }
}

__global__ void k_finalize(const float* lc, const float* lr, float* out, long N) {
  if (blockIdx.x == 0 && threadIdx.x == 0) {
    float cla = lc[0] / (float)((double)KPOS * (double)N);
    float s = 0.f;
    for (int i = 0; i < KPOS; ++i) s += lr[i];
    float reg = s / (float)KPOS;
    out[0] = cla + reg;  // total
    out[1] = cla;        // loss_cla
    out[2] = reg;        // loss_reg
  }
}

// ---------------------------------------------------------------- launcher
extern "C" void kernel_launch(void* const* d_in, const int* in_sizes, int n_in,
                              void* d_out, int out_size, void* d_ws, size_t ws_size,
                              hipStream_t stream) {
  const float* x      = (const float*)d_in[0];
  const int*   eidx   = (const int*)  d_in[1];
  const float* ea     = (const float*)d_in[2];
  const int*   rev    = (const int*)  d_in[3];
  const float* labels = (const float*)d_in[4];
  const float* Wi1 = (const float*)d_in[5];  const float* bi1 = (const float*)d_in[6];
  const float* Wh1 = (const float*)d_in[7];  const float* bh1 = (const float*)d_in[8];
  const float* Wo1 = (const float*)d_in[9];  const float* bo1 = (const float*)d_in[10];
  const float* Wi3 = (const float*)d_in[11]; const float* bi3 = (const float*)d_in[12];
  const float* Wh3 = (const float*)d_in[13]; const float* bh3 = (const float*)d_in[14];
  const float* Wo3 = (const float*)d_in[15]; const float* bo3 = (const float*)d_in[16];
  const float* Wg  = (const float*)d_in[17]; const float* bg  = (const float*)d_in[18];
  const float* Wc1 = (const float*)d_in[19]; const float* bc1 = (const float*)d_in[20];
  const float* Wc2 = (const float*)d_in[21]; const float* bc2 = (const float*)d_in[22];
  const float* Wr1 = (const float*)d_in[23]; const float* br1 = (const float*)d_in[24];
  const float* Wr2 = (const float*)d_in[25]; const float* br2 = (const float*)d_in[26];
  const float* pka_std = (const float*)d_in[28];

  const long N = in_sizes[4];          // 30000 nodes
  const long E = in_sizes[3];          // 120000 directed edges
  const int* src = eidx;
  const int* dst = eidx + E;
  const float ratio = (float)(((double)N - 1.0) / (1.0 + 1e-6));

  float* out = (float*)d_out;
  float* out_logits = out;             // [N,2]
  float* out_pka    = out + 2 * N;     // [N]
  float* out_scal   = out + 3 * N;     // total, loss_cla, loss_reg

  // ---- bump allocator over d_ws
  char* base = (char*)d_ws;
  size_t off = 0;
  auto alloc = [&](size_t bytes) -> void* {
    void* p = base + off;
    off = (off + bytes + 255) & ~(size_t)255;
    return p;
  };

  __bf16* Wi1p = (__bf16*)alloc((size_t)KP_H  * NP_H * 2);
  __bf16* Wh1p = (__bf16*)alloc((size_t)KP_H  * NP_H * 2);
  __bf16* Wo1p = (__bf16*)alloc((size_t)KP_CAT* NP_H * 2);
  __bf16* Wi3p = (__bf16*)alloc((size_t)KP_H  * NP_H * 2);
  __bf16* Wh3p = (__bf16*)alloc((size_t)KP_H  * NP_H * 2);
  __bf16* Wo3p = (__bf16*)alloc((size_t)KP_CAT* NP_H * 2);
  __bf16* Wc1p = (__bf16*)alloc((size_t)KP_H  * NP_C * 2);
  __bf16* Wr1p = (__bf16*)alloc((size_t)KP_H  * NP_C * 2);

  size_t apElems = (size_t)E * KP_H;
  if ((size_t)N * KP_CAT > apElems) apElems = (size_t)N * KP_CAT;
  __bf16* Ap = (__bf16*)alloc(apElems * 2);

  float* h0  = (float*)alloc((size_t)E * HDIM * 4);
  float* hA  = (float*)alloc((size_t)E * HDIM * 4);
  float* hB  = (float*)alloc((size_t)E * HDIM * 4);
  float* agg = (float*)alloc((size_t)N * HDIM * 4);
  float* h_static = (float*)alloc((size_t)N * HDIM * 4);
  float* h_cur    = (float*)alloc((size_t)N * HDIM * 4);
  float* h_in     = (float*)alloc((size_t)N * HDIM * 4);
  float* f1       = (float*)alloc((size_t)N * 128 * 4);
  float* scal     = (float*)alloc(64 * 4);   // [0]=lc_accum, [1..8]=lr
  int*   idxs     = (int*)  alloc(64);

  auto cdiv = [](long a, long b) { return (a + b - 1) / b; };

  auto packw = [&](const float* W, __bf16* Wp, int K, int Nw, int Kp, int Np) {
    long tot = (long)Kp * Np;
    k_pack_w<<<dim3((unsigned)cdiv(tot, 256)), dim3(256), 0, stream>>>(Wp, W, K, Nw, Kp, Np);
  };
  auto gemm = [&](const __bf16* A, int lda, const __bf16* B, int ldb,
                  const float* bias, const float* add, float* C,
                  long M, int Kp, int Nout, int relu) {
    dim3 g((unsigned)cdiv(M, 64), (unsigned)cdiv(Nout, 64));
    k_gemm<<<g, 128, 0, stream>>>(A, lda, B, ldb, bias, add, C, (int)M, Kp, Nout, relu);
  };
  auto zero = [&](float* p, long n) {
    k_zero_f<<<dim3((unsigned)cdiv(n, 256)), dim3(256), 0, stream>>>(p, n);
  };

  // One ChemProp DMPNN pass: xin [N,300] -> outv [N,300]
  auto dmpnn = [&](const float* xin,
                   const __bf16* Wip, const float* bi,
                   const __bf16* Whp, const float* bh,
                   const __bf16* Wop, const float* bo,
                   float* outv) {
    long totE = E * KP_H;
    k_pack_edge<<<dim3((unsigned)cdiv(totE, 256)), dim3(256), 0, stream>>>(Ap, xin, ea, src, E);
    gemm(Ap, KP_H, Wip, NP_H, bi, nullptr, h0, E, KP_H, HDIM, 1);
    const float* hprev = h0;
    float* houts[2] = {hA, hB};
    for (int it = 0; it < 3; ++it) {             // DEPTH-1 message steps
      float* hout = houts[it & 1];
      zero(agg, N * HDIM);
      k_scatter_add<<<dim3((unsigned)cdiv(E * HDIM, 256)), dim3(256), 0, stream>>>(agg, hprev, dst, E);
      k_pack_msg<<<dim3((unsigned)cdiv(totE, 256)), dim3(256), 0, stream>>>(Ap, agg, hprev, src, rev, E);
      gemm(Ap, KP_H, Whp, NP_H, bh, h0, hout, E, KP_H, HDIM, 1);
      hprev = hout;
    }
    zero(agg, N * HDIM);
    k_scatter_add<<<dim3((unsigned)cdiv(E * HDIM, 256)), dim3(256), 0, stream>>>(agg, hprev, dst, E);
    long totC = N * KP_CAT;
    k_pack_cat2<<<dim3((unsigned)cdiv(totC, 256)), dim3(256), 0, stream>>>(Ap, xin, agg, N);
    gemm(Ap, KP_CAT, Wop, NP_H, bo, nullptr, outv, N, KP_CAT, HDIM, 1);
  };

  // ---- pack all weights to padded bf16
  packw(Wi1, Wi1p, HDIM + BONDD, HDIM, KP_H,   NP_H);
  packw(Wh1, Wh1p, HDIM,         HDIM, KP_H,   NP_H);
  packw(Wo1, Wo1p, 2 * HDIM,     HDIM, KP_CAT, NP_H);
  packw(Wi3, Wi3p, HDIM + BONDD, HDIM, KP_H,   NP_H);
  packw(Wh3, Wh3p, HDIM,         HDIM, KP_H,   NP_H);
  packw(Wo3, Wo3p, 2 * HDIM,     HDIM, KP_CAT, NP_H);
  packw(Wc1, Wc1p, HDIM, 128, KP_H, NP_C);
  packw(Wr1, Wr1p, HDIM, 128, KP_H, NP_C);

  // ---- two static DMPNN passes
  dmpnn(x,        Wi1p, bi1, Wh1p, bh1, Wo1p, bo1, h_static);
  dmpnn(h_static, Wi3p, bi3, Wh3p, bh3, Wo3p, bo3, h_cur);

  // ---- top-K labelled atoms sorted ascending; zero loss accumulators
  k_topk<<<1, 256, 0, stream>>>(labels, N, idxs);
  zero(scal, 16);

  // ---- sequential scan over K positives
  for (int step = 0; step < KPOS; ++step) {
    long totP = N * KP_H;
    k_pack_plain<<<dim3((unsigned)cdiv(totP, 256)), dim3(256), 0, stream>>>(Ap, h_cur, N);
    gemm(Ap, KP_H, Wc1p, NP_C, bc1, nullptr, f1, N, KP_H, 128, 1);
    k_ce<<<dim3((unsigned)cdiv(N, 256)), dim3(256), 0, stream>>>(
        f1, Wc2, bc2, idxs, step, ratio, N, &scal[0],
        (step == KPOS - 1) ? out_logits : nullptr);
    k_reg_row<<<1, 128, 0, stream>>>(h_cur, idxs, step, Wr1, br1, Wr2, br2,
                                     labels, pka_std, &scal[1]);
    k_copy_f<<<dim3((unsigned)cdiv(N * HDIM, 256)), dim3(256), 0, stream>>>(h_in, h_static, N * HDIM);
    k_gate_row<<<1, 320, 0, stream>>>(h_in, h_cur, idxs, step, Wg, bg);
    dmpnn(h_in, Wi3p, bi3, Wh3p, bh3, Wo3p, bo3, h_cur);
  }

  // ---- final pKa regression over all nodes
  long totP = N * KP_H;
  k_pack_plain<<<dim3((unsigned)cdiv(totP, 256)), dim3(256), 0, stream>>>(Ap, h_cur, N);
  gemm(Ap, KP_H, Wr1p, NP_C, br1, nullptr, f1, N, KP_H, 128, 1);
  k_pka<<<dim3((unsigned)cdiv(N, 256)), dim3(256), 0, stream>>>(f1, Wr2, br2, out_pka, N);

  // ---- scalar losses
  k_finalize<<<1, 64, 0, stream>>>(&scal[0], &scal[1], out_scal, N);
}
